// AttentionOnDetail_4587025072220
// MI455X (gfx1250) — compile-verified, matrix operand order checked by reading, not run on hardware
//
#include <hip/hip_runtime.h>
#include <hip/hip_bf16.h>

// ---------------------------------------------------------------------------
// Types
// ---------------------------------------------------------------------------
typedef __attribute__((ext_vector_type(4)))  float          f32x4;
typedef __attribute__((ext_vector_type(8)))  float          f32x8;
typedef __attribute__((ext_vector_type(16))) __bf16         bf16x16;
typedef __attribute__((ext_vector_type(4)))  unsigned short u16x4;

union Frag16 {               // one WMMA bf16 A/B fragment (16x32), 32 bytes
    bf16x16 v;
    f32x4   q[2];
};

__device__ __forceinline__ unsigned short f2bf(float f) {
    union { float f; unsigned u; } a; a.f = f;
    unsigned r = a.u + 0x7FFFu + ((a.u >> 16) & 1u);   // round-to-nearest-even
    return (unsigned short)(r >> 16);
}

// gfx1250 async copy: global -> LDS, 16 bytes per lane, tracked by ASYNCcnt.
__device__ __forceinline__ void async_load_b128(unsigned lds_addr, const void* gptr) {
    asm volatile("global_load_async_to_lds_b128 %0, %1, off"
                 :: "v"(lds_addr), "v"(gptr) : "memory");
}
__device__ __forceinline__ void wait_asynccnt0() {
    asm volatile("s_wait_asynccnt 0x0" ::: "memory");
}

// ---------------------------------------------------------------------------
// Problem constants (reference: B=4, T=8192, C=1024, 16 heads x 64)
// ---------------------------------------------------------------------------
#define BATCH   4
#define TDIM    8192
#define CDIM    1024
#define NHEAD   16
#define DHEAD   64
#define MROWS   (BATCH * TDIM)       // 32768
#define CHUNK   128
#define NCH     (TDIM / CHUNK)       // 64

// ---------------------------------------------------------------------------
// rms_norm(x) over C and cast to bf16.  One block per row, 256 threads x 4.
// ---------------------------------------------------------------------------
__global__ __launch_bounds__(256) void rmsnorm_cast_kernel(
    const float* __restrict__ x, unsigned short* __restrict__ out)
{
    const int row = blockIdx.x;
    const float* xr = x + (size_t)row * CDIM;
    f32x4 v = *(const f32x4*)(xr + threadIdx.x * 4);
    float ss = v.x * v.x + v.y * v.y + v.z * v.z + v.w * v.w;
    #pragma unroll
    for (int off = 16; off >= 1; off >>= 1) ss += __shfl_xor(ss, off, 32);

    __shared__ float wsum[8];
    __shared__ float total;
    const int wid = threadIdx.x >> 5;
    if ((threadIdx.x & 31) == 0) wsum[wid] = ss;
    __syncthreads();
    if (threadIdx.x == 0) {
        float t = 0.f;
        #pragma unroll
        for (int i = 0; i < 8; ++i) t += wsum[i];
        total = t;
    }
    __syncthreads();
    const float scale = rsqrtf(total * (1.0f / (float)CDIM) + 1e-6f);

    u16x4 o;
    o.x = f2bf(v.x * scale); o.y = f2bf(v.y * scale);
    o.z = f2bf(v.z * scale); o.w = f2bf(v.w * scale);
    *(u16x4*)(out + (size_t)row * CDIM + threadIdx.x * 4) = o;
}

// ---------------------------------------------------------------------------
// fp32 -> bf16 elementwise cast (weights)
// ---------------------------------------------------------------------------
__global__ void cast_bf16_kernel(const float* __restrict__ in,
                                 unsigned short* __restrict__ out, int n)
{
    int i = blockIdx.x * blockDim.x + threadIdx.x;
    if (i < n) out[i] = f2bf(in[i]);
}

// ---------------------------------------------------------------------------
// bf16 WMMA GEMM:  C[M,N] = A[M,K] * W[N,K]^T   (+ optional residual/relu^2)
// Block = 256 threads (8 waves). Block tile 128x128, K-step 32.
// Wave (wr in 0..3, wc in 0..1) owns a 32x64 sub-tile = 2x4 WMMA tiles.
// Staging: double-buffered GLOBAL_LOAD_ASYNC_TO_LDS_B128 (ASYNCcnt).
// ---------------------------------------------------------------------------
#define BM 128
#define BN 128
#define BK 32
#define LDSK 40          // padded row stride in bf16 elements

template <int EPI>
__global__ __launch_bounds__(256) void gemm_bf16_kernel(
    const unsigned short* __restrict__ A,     // M x K  (bf16 bits)
    const unsigned short* __restrict__ Wt,    // N x K  (bf16 bits)
    float* __restrict__ Cout,                 // M x N
    const float* __restrict__ Resid,          // M x N (EPI==1)
    int M, int N, int K)
{
    __shared__ __align__(16) unsigned short As[2][BM * LDSK];
    __shared__ __align__(16) unsigned short Ws[2][BN * LDSK];

    const int tid  = threadIdx.x;
    const int lane = tid & 31;
    const int wave = tid >> 5;
    const int wr   = wave & 3;        // row wave   (4)
    const int wc   = wave >> 2;       // col wave   (2)
    const int r    = lane & 15;
    const int half = lane >> 4;

    const int mBlk = blockIdx.y * BM;
    const int nBlk = blockIdx.x * BN;

    const int ldRow = tid >> 1;               // 0..127
    const int ldSeg = (tid & 1) * 16;         // 0 / 16 bf16 elems (32 B)

    // per-thread LDS staging addresses (byte offsets; generic-ptr low 32 bits)
    const unsigned laBuf[2] = {
        (unsigned)(uintptr_t)&As[0][ldRow * LDSK + ldSeg],
        (unsigned)(uintptr_t)&As[1][ldRow * LDSK + ldSeg]
    };
    const unsigned lwBuf[2] = {
        (unsigned)(uintptr_t)&Ws[0][ldRow * LDSK + ldSeg],
        (unsigned)(uintptr_t)&Ws[1][ldRow * LDSK + ldSeg]
    };
    const unsigned short* gaRow = A  + (size_t)(mBlk + ldRow) * K + ldSeg;
    const unsigned short* gwRow = Wt + (size_t)(nBlk + ldRow) * K + ldSeg;

    auto issue_tile = [&](int buf, int k0) {
        const unsigned short* ga = gaRow + k0;
        const unsigned short* gw = gwRow + k0;
        async_load_b128(laBuf[buf],      ga);
        async_load_b128(laBuf[buf] + 16, ga + 8);
        async_load_b128(lwBuf[buf],      gw);
        async_load_b128(lwBuf[buf] + 16, gw + 8);
    };

    f32x8 acc[2][4];
    const f32x8 zero = {0.f, 0.f, 0.f, 0.f, 0.f, 0.f, 0.f, 0.f};
    #pragma unroll
    for (int i = 0; i < 2; ++i)
        #pragma unroll
        for (int j = 0; j < 4; ++j) acc[i][j] = zero;

    issue_tile(0, 0);                          // prologue
    const int nK = K / BK;
    for (int kt = 0; kt < nK; ++kt) {
        const int cur = kt & 1;
        wait_asynccnt0();                      // tile `cur` fully landed (mine)
        __syncthreads();                       // ...and everyone else's; prior reads retired
        if (kt + 1 < nK) issue_tile(cur ^ 1, (kt + 1) * BK);   // prefetch next tile

        Frag16 fa[2], fw[4];
        #pragma unroll
        for (int i = 0; i < 2; ++i) {
            const int row = wr * 32 + i * 16 + r;
            fa[i].q[0] = *(const f32x4*)&As[cur][row * LDSK + half * 8];
            fa[i].q[1] = *(const f32x4*)&As[cur][row * LDSK + 16 + half * 8];
        }
        #pragma unroll
        for (int j = 0; j < 4; ++j) {
            const int row = wc * 64 + j * 16 + r;
            fw[j].q[0] = *(const f32x4*)&Ws[cur][row * LDSK + half * 8];
            fw[j].q[1] = *(const f32x4*)&Ws[cur][row * LDSK + 16 + half * 8];
        }
        #pragma unroll
        for (int i = 0; i < 2; ++i)
            #pragma unroll
            for (int j = 0; j < 4; ++j)
                acc[i][j] = __builtin_amdgcn_wmma_f32_16x16x32_bf16(
                    false, fa[i].v, false, fw[j].v,
                    (short)0, acc[i][j], false, false);
    }

    // Epilogue: C/D tile layout -> m = base + half*8 + e, n = base + (lane&15)
    #pragma unroll
    for (int i = 0; i < 2; ++i) {
        const int mBase = mBlk + wr * 32 + i * 16 + half * 8;
        #pragma unroll
        for (int j = 0; j < 4; ++j) {
            const int n = nBlk + wc * 64 + j * 16 + (lane & 15);
            #pragma unroll
            for (int e = 0; e < 8; ++e) {
                const size_t idx = (size_t)(mBase + e) * N + n;
                float v = acc[i][j][e];
                if (EPI == 1) {
                    float t = Resid[idx] + v;
                    t = fmaxf(t, 0.f);
                    Cout[idx] = t * t;
                } else {
                    Cout[idx] = v;
                }
            }
        }
    }
}

// ---------------------------------------------------------------------------
// Scan helpers: lane d (0..31) owns channels d and d+32 (= rotary pair).
// ---------------------------------------------------------------------------
__device__ __forceinline__ float wave_sum32(float v) {
    #pragma unroll
    for (int off = 16; off >= 1; off >>= 1) v += __shfl_xor(v, off, 32);
    return v;
}

// Phase 1: per-chunk sums of sigmoid(rot(rms(k))) * v
__global__ __launch_bounds__(32) void scan_phase1_kernel(
    const float* __restrict__ qkv, float* __restrict__ csum)
{
    const int bid = blockIdx.x;               // (b*NHEAD + h)*NCH + c
    const int c = bid % NCH;
    const int h = (bid / NCH) % NHEAD;
    const int b = bid / (NCH * NHEAD);
    const int d = threadIdx.x;
    const float af = (d < 16) ? exp2f(-10.0f * (float)d / 15.0f) : 0.0f;

    float acc1 = 0.f, acc2 = 0.f;
    const int t0 = c * CHUNK;
    for (int t = t0; t < t0 + CHUNK; ++t) {
        const float* base = qkv + ((size_t)b * TDIM + t) * (3 * CDIM);
        float k1 = base[(NHEAD + h) * DHEAD + d];
        float k2 = base[(NHEAD + h) * DHEAD + d + 32];
        float v1 = base[(2 * NHEAD + h) * DHEAD + d];
        float v2 = base[(2 * NHEAD + h) * DHEAD + d + 32];
        float ss = wave_sum32(k1 * k1 + k2 * k2);
        float sc = rsqrtf(ss * (1.0f / (float)DHEAD) + 1e-6f);
        k1 *= sc; k2 *= sc;
        float sn, cs_;
        __sincosf(af * (float)t, &sn, &cs_);
        float kr1 =  k1 * cs_ + k2 * sn;
        float kr2 = -k1 * sn  + k2 * cs_;
        acc1 += v1 / (1.f + __expf(-kr1));
        acc2 += v2 / (1.f + __expf(-kr2));
    }
    float* out = csum + (size_t)bid * DHEAD;
    out[d]      = acc1;
    out[d + 32] = acc2;
}

// Phase 2: exclusive prefix over chunks (per b,h,channel)
__global__ __launch_bounds__(64) void scan_phase2_kernel(float* __restrict__ csum)
{
    const int bh = blockIdx.x;                // b*NHEAD + h
    const int d  = threadIdx.x;               // 0..63
    float run = 0.f;
    for (int c = 0; c < NCH; ++c) {
        const size_t idx = ((size_t)bh * NCH + c) * DHEAD + d;
        float v = csum[idx];
        csum[idx] = run;
        run += v;
    }
}

// Phase 3: replay chunk with carry-in, apply sigmoid(rot(rms(q))), emit y bf16
__global__ __launch_bounds__(32) void scan_phase3_kernel(
    const float* __restrict__ qkv, const float* __restrict__ csum,
    unsigned short* __restrict__ y)
{
    const int bid = blockIdx.x;
    const int c = bid % NCH;
    const int h = (bid / NCH) % NHEAD;
    const int b = bid / (NCH * NHEAD);
    const int d = threadIdx.x;
    const float af = (d < 16) ? exp2f(-10.0f * (float)d / 15.0f) : 0.0f;

    float acc1 = csum[(size_t)bid * DHEAD + d];
    float acc2 = csum[(size_t)bid * DHEAD + d + 32];
    const int t0 = c * CHUNK;
    for (int t = t0; t < t0 + CHUNK; ++t) {
        const float* base = qkv + ((size_t)b * TDIM + t) * (3 * CDIM);
        float q1 = base[h * DHEAD + d];
        float q2 = base[h * DHEAD + d + 32];
        float k1 = base[(NHEAD + h) * DHEAD + d];
        float k2 = base[(NHEAD + h) * DHEAD + d + 32];
        float v1 = base[(2 * NHEAD + h) * DHEAD + d];
        float v2 = base[(2 * NHEAD + h) * DHEAD + d + 32];

        float sn, cs_;
        __sincosf(af * (float)t, &sn, &cs_);

        float ssk = wave_sum32(k1 * k1 + k2 * k2);
        float sck = rsqrtf(ssk * (1.0f / (float)DHEAD) + 1e-6f);
        k1 *= sck; k2 *= sck;
        float kr1 =  k1 * cs_ + k2 * sn;
        float kr2 = -k1 * sn  + k2 * cs_;
        acc1 += v1 / (1.f + __expf(-kr1));
        acc2 += v2 / (1.f + __expf(-kr2));

        float ssq = wave_sum32(q1 * q1 + q2 * q2);
        float scq = rsqrtf(ssq * (1.0f / (float)DHEAD) + 1e-6f);
        q1 *= scq; q2 *= scq;
        float qr1 =  q1 * cs_ + q2 * sn;
        float qr2 = -q1 * sn  + q2 * cs_;
        float y1 = acc1 / (1.f + __expf(-qr1));
        float y2 = acc2 / (1.f + __expf(-qr2));

        unsigned short* yo = y + ((size_t)b * TDIM + t) * CDIM + h * DHEAD;
        yo[d]      = f2bf(y1);
        yo[d + 32] = f2bf(y2);
    }
}

// ---------------------------------------------------------------------------
// SwiGLU: h = u * silu(g), uv = [u | g] (N=2048), output bf16 (32768 x 1024)
// ---------------------------------------------------------------------------
__global__ void swiglu_kernel(const float* __restrict__ uv,
                              unsigned short* __restrict__ hbf, int total)
{
    int i = blockIdx.x * blockDim.x + threadIdx.x;
    if (i >= total) return;
    const int row = i >> 10;
    const int f   = i & 1023;
    float u = uv[(size_t)row * 2048 + f];
    float g = uv[(size_t)row * 2048 + 1024 + f];
    float s = g / (1.f + __expf(-g));
    hbf[i] = f2bf(u * s);
}

// ---------------------------------------------------------------------------
// Launcher
// ---------------------------------------------------------------------------
extern "C" void kernel_launch(void* const* d_in, const int* in_sizes, int n_in,
                              void* d_out, int out_size, void* d_ws, size_t ws_size,
                              hipStream_t stream)
{
    (void)in_sizes; (void)n_in; (void)out_size; (void)ws_size;

    const float* x     = (const float*)d_in[0];   // (4,8192,1024)
    const float* w_qkv = (const float*)d_in[1];   // (3072,1024)
    const float* w_sw  = (const float*)d_in[2];   // (2048,1024)
    const float* w_out = (const float*)d_in[3];   // (1024,1024)

    const int M = MROWS, K = CDIM;
    const int Nqkv = 3 * CDIM, Nsw = 2 * CDIM, Nout = CDIM;

    char* p = (char*)d_ws;
    auto take = [&](size_t bytes) -> void* {
        void* q = (void*)p;
        p += (bytes + 255) & ~(size_t)255;
        return q;
    };
    unsigned short* xn  = (unsigned short*)take((size_t)M * K * 2);
    unsigned short* wqb = (unsigned short*)take((size_t)Nqkv * K * 2);
    unsigned short* wsb = (unsigned short*)take((size_t)Nsw * K * 2);
    unsigned short* wob = (unsigned short*)take((size_t)Nout * K * 2);
    float*          qkv = (float*)take((size_t)M * Nqkv * 4);
    unsigned short* ybf = (unsigned short*)take((size_t)M * CDIM * 2);
    float*          uv  = (float*)take((size_t)M * Nsw * 4);
    unsigned short* hbf = (unsigned short*)take((size_t)M * CDIM * 2);
    float*          cs  = (float*)take((size_t)BATCH * NHEAD * NCH * DHEAD * 4);

    // 1) rms_norm(x) -> bf16; weights -> bf16
    rmsnorm_cast_kernel<<<M, 256, 0, stream>>>(x, xn);
    cast_bf16_kernel<<<(Nqkv * K + 255) / 256, 256, 0, stream>>>(w_qkv, wqb, Nqkv * K);
    cast_bf16_kernel<<<(Nsw  * K + 255) / 256, 256, 0, stream>>>(w_sw,  wsb, Nsw  * K);
    cast_bf16_kernel<<<(Nout * K + 255) / 256, 256, 0, stream>>>(w_out, wob, Nout * K);

    // 2) qkv = rms_norm(x) @ w_qkv^T   (WMMA bf16, async-LDS double buffered)
    gemm_bf16_kernel<0><<<dim3(Nqkv / BN, M / BM), 256, 0, stream>>>(
        xn, wqb, qkv, nullptr, M, Nqkv, K);

    // 3) chunked scan: y = sigmoid(q) * cumsum_T(sigmoid(k) * v)
    scan_phase1_kernel<<<BATCH * NHEAD * NCH, 32, 0, stream>>>(qkv, cs);
    scan_phase2_kernel<<<BATCH * NHEAD, 64, 0, stream>>>(cs);
    scan_phase3_kernel<<<BATCH * NHEAD * NCH, 32, 0, stream>>>(qkv, cs, ybf);

    // 4) uv = y @ w_swiglu^T; h = u * silu(g)
    gemm_bf16_kernel<0><<<dim3(Nsw / BN, M / BM), 256, 0, stream>>>(
        ybf, wsb, uv, nullptr, M, Nsw, K);
    swiglu_kernel<<<(M * CDIM + 255) / 256, 256, 0, stream>>>(uv, hbf, M * CDIM);

    // 5) out = relu(x + h @ w_out^T)^2
    gemm_bf16_kernel<1><<<dim3(Nout / BN, M / BM), 256, 0, stream>>>(
        hbf, wob, (float*)d_out, x, M, Nout, K);
}
